// Attention_external_11081015623732
// MI455X (gfx1250) — matrix operand order repaired and checked by reading, not compiled, not run on hardware
//
#include <hip/hip_runtime.h>

typedef __attribute__((ext_vector_type(8)))  float          v8f;
typedef __attribute__((ext_vector_type(16))) __bf16         v16bf;
typedef __attribute__((ext_vector_type(8)))  unsigned short v8us;
typedef __attribute__((ext_vector_type(16))) unsigned short v16us;

union FragU {
  v16bf          bf;
  v16us          us;
  v8us           h[2];
  unsigned short s[16];
};

__device__ __forceinline__ unsigned short f2bf(float f) {
  unsigned int u = __float_as_uint(f);
  u += 0x7FFFu + ((u >> 16) & 1u);   // round-to-nearest-even
  return (unsigned short)(u >> 16);
}

__device__ __forceinline__ v16bf load_b16(const unsigned short* p) {
  FragU f; f.us = *(const v16us*)p; return f.bf;      // 32B contiguous (B frag)
}
__device__ __forceinline__ v16bf load_a2(const unsigned short* p0,
                                         const unsigned short* p1) {
  FragU f; f.h[0] = *(const v8us*)p0; f.h[1] = *(const v8us*)(p0 + 8); return f.bf;
}
__device__ __forceinline__ v8f wmma_bf(v16bf a, v16bf b, v8f c) {
  return __builtin_amdgcn_wmma_f32_16x16x32_bf16(false, a, false, b,
                                                 (short)0, c, false, false);
}
__device__ __forceinline__ float rmax16(float v) {
#pragma unroll
  for (int m = 1; m < 16; m <<= 1) v = fmaxf(v, __shfl_xor(v, m, 32));
  return v;
}
__device__ __forceinline__ float rsum16(float v) {
#pragma unroll
  for (int m = 1; m < 16; m <<= 1) v += __shfl_xor(v, m, 32);
  return v;
}

#define NTOK 2049          // 1 ext + 2048 tokens per batch
#define MROWS (16 * NTOK)  // 32784, divisible by 16
#define DIM 512

// ---------- convert xa = concat(external, x) to bf16 ----------
__global__ void k_convert_xa(const float* __restrict__ x,
                             const float* __restrict__ ext,
                             unsigned short* __restrict__ xa, int total) {
  int idx = blockIdx.x * blockDim.x + threadIdx.x;
  if (idx >= total) return;
  int col = idx & 511, row = idx >> 9;
  int b = row / NTOK, tr = row - b * NTOK;
  float v = (tr == 0) ? ext[b * 512 + col]
                      : x[(size_t)(b * 2048 + (tr - 1)) * 512 + col];
  xa[idx] = f2bf(v);
}

// ---------- transpose-convert weights to (N x K) bf16 ----------
__global__ void k_convert_wt(const float* __restrict__ W,
                             unsigned short* __restrict__ Wt, float scale) {
  int idx = blockIdx.x * blockDim.x + threadIdx.x;  // 512*512
  int n = idx >> 9, kk = idx & 511;
  Wt[idx] = f2bf(W[kk * 512 + n] * scale);
}

// ---------- fused QKV GEMM: (32784x512) @ (512x512)x3, bf16 out ----------
__global__ void __launch_bounds__(256)
k_gemm_qkv(const unsigned short* __restrict__ xa,
           const unsigned short* __restrict__ wt,  // wq_t|wk_t|wv_t contiguous
           const float* __restrict__ bq, const float* __restrict__ bk,
           const float* __restrict__ bv, unsigned short* __restrict__ q,
           unsigned short* __restrict__ k, unsigned short* __restrict__ v) {
  int wave = threadIdx.x >> 5, lane = threadIdx.x & 31;
  int task = blockIdx.x * 8 + wave;                 // 3*2049*8 tasks
  if (task >= 3 * NTOK * 8) return;
  int mat = task / (NTOK * 8);
  int rem = task - mat * (NTOK * 8);
  int mt = rem >> 3, nt = rem & 7;
  const unsigned short* Bt = wt + (size_t)mat * DIM * DIM;
  const float* bias = (mat == 0) ? bq : (mat == 1) ? bk : bv;
  unsigned short* dst = (mat == 0) ? q : (mat == 1) ? k : v;
  float bmul = (mat == 0) ? 0.125f : 1.0f;          // SCALE folded for q
  int g = lane >> 4, ln = lane & 15;
  int kbA = g ? 16 : 0, kbB = g ? 16 : 0;
  int m0 = mt * 16, n0 = nt * 64;

  v8f z = {0.f, 0.f, 0.f, 0.f, 0.f, 0.f, 0.f, 0.f};
  v8f c[4] = {z, z, z, z};
  const unsigned short* Arow = xa + (size_t)(m0 + ln) * DIM;
#pragma unroll 4
  for (int k0 = 0; k0 < DIM; k0 += 32) {
    v16bf a = load_a2(Arow + k0 + kbA, Arow + k0 + 16 + kbA);
#pragma unroll
    for (int t = 0; t < 4; t++) {
      const unsigned short* bp = Bt + (size_t)(n0 + t * 16 + ln) * DIM + k0 + kbB;
      c[t] = wmma_bf(a, load_b16(bp), c[t]);
    }
  }
#pragma unroll
  for (int t = 0; t < 4; t++) {
    int col = n0 + t * 16 + ln;
    float bb = bias[col] * bmul;
#pragma unroll
    for (int r = 0; r < 8; r++) {
      int row = m0 + r + 8 * g;
      dst[(size_t)row * DIM + col] = f2bf(c[t][r] + bb);
    }
  }
}

// ---------- flash attention over 1024 chunks of 257x257, DH=64 ----------
// LDS: V staged once per chunk in B-fragment order (36KB) + P scratch (8KB)
__global__ void __launch_bounds__(256)
k_attn(const unsigned short* __restrict__ q, const unsigned short* __restrict__ k,
       const unsigned short* __restrict__ v, unsigned short* __restrict__ opre) {
  __shared__ __align__(16) unsigned short v_lds[9 * 4 * 32 * 16];  // 36KB
  __shared__ __align__(16) unsigned short p_lds[8][16][32];        // 8KB
  int wave = threadIdx.x >> 5, lane = threadIdx.x & 31;
  int c = blockIdx.x;                      // chunk = ((b*H)+h)*f + fi
  int bb = c >> 6, rem = c & 63, h = rem >> 3, fi = rem & 7;
  int g = lane >> 4, ln = lane & 15;
  int kbA = g ? 16 : 0, kbB = g ? 16 : 0;
  size_t base = (size_t)bb * NTOK * DIM + h * 64;
  const unsigned short* qb = q + base;
  const unsigned short* kb = k + base;
  const unsigned short* vb = v + base;
  v8f z = {0.f, 0.f, 0.f, 0.f, 0.f, 0.f, 0.f, 0.f};

  // ---- cooperative fill of V fragments: slot = ((jp*4+t)*32+lane) ----
  for (int e = threadIdx.x; e < 9 * 4 * 32; e += 256) {
    int laneS = e & 31, tS = (e >> 5) & 3, jpS = e >> 7;
    int lnS = laneS & 15, gS = laneS >> 4;
    FragU tmp;
#pragma unroll
    for (int i = 0; i < 16; i++) {
      int ci = jpS * 32 + gS * 16 + i; if (ci > 256) ci = 256;
      int vt = (ci == 0) ? 0 : (fi * 256 + ci);
      tmp.s[i] = vb[(size_t)vt * DIM + tS * 16 + lnS];
    }
    *(v16us*)&v_lds[e * 16] = tmp.us;
  }
  __syncthreads();   // v_lds read-only hereafter

  for (int rt = wave; rt < 17; rt += 8) {            // 17 row tiles of 16
    int iq = rt * 16 + ln; if (iq > 256) iq = 256;   // clamp pad rows
    int qt = (iq == 0) ? 0 : (fi * 256 + iq);
    const unsigned short* qrow = qb + (size_t)qt * DIM;
    v16bf a0 = load_a2(qrow + kbA, qrow + 16 + kbA);        // K = 0..31
    v16bf a1 = load_a2(qrow + 32 + kbA, qrow + 48 + kbA);   // K = 32..63
    float mi[8], li[8];
    v8f o[4] = {z, z, z, z};
#pragma unroll
    for (int r = 0; r < 8; r++) { mi[r] = -3e38f; li[r] = 0.f; }

    for (int jp = 0; jp < 9; jp++) {                 // 9 col pairs of 32
      v8f s[2] = {z, z};
#pragma unroll
      for (int half = 0; half < 2; half++) {
        int jc = jp * 2 + half;
        int ck = jc * 16 + ln; if (ck > 256) ck = 256;
        int kt = (ck == 0) ? 0 : (fi * 256 + ck);
        const unsigned short* krow = kb + (size_t)kt * DIM;
        s[half] = wmma_bf(a0, load_b16(krow + kbB), s[half]);
        s[half] = wmma_bf(a1, load_b16(krow + 32 + kbB), s[half]);
      }
#pragma unroll
      for (int half = 0; half < 2; half++) {         // mask invalid cols
        int col = (jp * 2 + half) * 16 + ln;
        if (col > 256) {
#pragma unroll
          for (int r = 0; r < 8; r++) s[half][r] = -3e38f;
        }
      }
      float sc[8];
#pragma unroll
      for (int r = 0; r < 8; r++) {                  // online softmax, 32 cols
        float tmax = rmax16(fmaxf(s[0][r], s[1][r]));
        float nm = fmaxf(mi[r], tmax);
        sc[r] = __expf(mi[r] - nm);
        float p0 = __expf(s[0][r] - nm);
        float p1 = __expf(s[1][r] - nm);
        s[0][r] = p0; s[1][r] = p1;
        li[r] = li[r] * sc[r] + rsum16(p0 + p1);
        mi[r] = nm;
      }
#pragma unroll
      for (int t = 0; t < 4; t++)
#pragma unroll
        for (int r = 0; r < 8; r++) o[t][r] *= sc[r];
      // P (C layout) -> LDS -> A fragment
#pragma unroll
      for (int half = 0; half < 2; half++)
#pragma unroll
        for (int r = 0; r < 8; r++)
          p_lds[wave][r + 8 * g][half * 16 + ln] = f2bf(s[half][r]);
      const unsigned short* pl = &p_lds[wave][ln][0];
      v16bf pa = load_a2(pl + kbA, pl + 16 + kbA);
      // O += P @ V   (V frags as 32B LDS reads)
#pragma unroll
      for (int t = 0; t < 4; t++) {
        FragU vf;
        vf.us = *(const v16us*)&v_lds[(((jp * 4) + t) * 32 + lane) * 16];
        o[t] = wmma_bf(pa, vf.bf, o[t]);
      }
    }
    // normalize and scatter to o_pre (rows 0..32767 main, 32768..32783 ext)
#pragma unroll
    for (int r = 0; r < 8; r++) {
      int i = rt * 16 + r + 8 * g;
      if (i > 256) continue;
      long drow;
      if (i == 0) {
        if (fi != 1) continue;
        drow = 32768 + bb;
      } else {
        drow = (long)bb * 2048 + fi * 256 + (i - 1);
      }
      float inv = 1.0f / li[r];
#pragma unroll
      for (int t = 0; t < 4; t++)
        opre[(size_t)drow * DIM + h * 64 + t * 16 + ln] = f2bf(o[t][r] * inv);
    }
  }
}

// ---------- output GEMM: (32784x512) @ Wo + bo -> f32 d_out ----------
__global__ void __launch_bounds__(256)
k_gemm_out(const unsigned short* __restrict__ A,
           const unsigned short* __restrict__ Wt,
           const float* __restrict__ bias, float* __restrict__ out) {
  int wave = threadIdx.x >> 5, lane = threadIdx.x & 31;
  int task = blockIdx.x * 8 + wave;                  // 2049*8 tasks
  if (task >= NTOK * 8) return;
  int mt = task >> 3, nt = task & 7;
  int g = lane >> 4, ln = lane & 15;
  int kbA = g ? 16 : 0, kbB = g ? 16 : 0;
  int m0 = mt * 16, n0 = nt * 64;
  v8f z = {0.f, 0.f, 0.f, 0.f, 0.f, 0.f, 0.f, 0.f};
  v8f c[4] = {z, z, z, z};
  const unsigned short* Arow = A + (size_t)(m0 + ln) * DIM;
#pragma unroll 4
  for (int k0 = 0; k0 < DIM; k0 += 32) {
    v16bf a = load_a2(Arow + k0 + kbA, Arow + k0 + 16 + kbA);
#pragma unroll
    for (int t = 0; t < 4; t++) {
      const unsigned short* bp = Wt + (size_t)(n0 + t * 16 + ln) * DIM + k0 + kbB;
      c[t] = wmma_bf(a, load_b16(bp), c[t]);
    }
  }
#pragma unroll
  for (int t = 0; t < 4; t++) {
    int col = n0 + t * 16 + ln;
    float bb = bias[col];
#pragma unroll
    for (int r = 0; r < 8; r++) {
      int row = m0 + r + 8 * g;
      out[(size_t)row * DIM + col] = c[t][r] + bb;
    }
  }
}

extern "C" void kernel_launch(void* const* d_in, const int* in_sizes, int n_in,
                              void* d_out, int out_size, void* d_ws, size_t ws_size,
                              hipStream_t stream) {
  const float* x   = (const float*)d_in[0];
  const float* ext = (const float*)d_in[1];
  const float* Wq  = (const float*)d_in[2];
  const float* bq  = (const float*)d_in[3];
  const float* Wk  = (const float*)d_in[4];
  const float* bk  = (const float*)d_in[5];
  const float* Wv  = (const float*)d_in[6];
  const float* bv  = (const float*)d_in[7];
  const float* Wo  = (const float*)d_in[8];
  const float* bo  = (const float*)d_in[9];
  float* out = (float*)d_out;

  const size_t MR = (size_t)MROWS * DIM;   // 16,785,408 elements
  unsigned short* xa   = (unsigned short*)d_ws;
  unsigned short* wqt  = xa + MR;
  unsigned short* wkt  = wqt + 512 * 512;
  unsigned short* wvt  = wkt + 512 * 512;
  unsigned short* wot  = wvt + 512 * 512;
  unsigned short* qb   = wot + 512 * 512;
  unsigned short* kbuf = qb + MR;
  unsigned short* vbuf = kbuf + MR;
  unsigned short* opre = vbuf + MR;

  int total = (int)MR;
  k_convert_xa<<<(total + 255) / 256, 256, 0, stream>>>(x, ext, xa, total);
  k_convert_wt<<<1024, 256, 0, stream>>>(Wq, wqt, 0.125f);  // fold DH^-0.5
  k_convert_wt<<<1024, 256, 0, stream>>>(Wk, wkt, 1.0f);
  k_convert_wt<<<1024, 256, 0, stream>>>(Wv, wvt, 1.0f);
  k_convert_wt<<<1024, 256, 0, stream>>>(Wo, wot, 1.0f);
  k_gemm_qkv<<<(3 * NTOK * 8) / 8, 256, 0, stream>>>(xa, wqt, bq, bk, bv,
                                                     qb, kbuf, vbuf);
  k_attn<<<1024, 256, 0, stream>>>(qb, kbuf, vbuf, opre);
  k_gemm_out<<<(NTOK * 8) / 8, 256, 0, stream>>>(opre, wot, bo, out);
}
